// GeometricEncoder_67121748902417
// MI455X (gfx1250) — matrix operand order repaired and checked by reading, not compiled, not run on hardware
//
#include <hip/hip_runtime.h>

// ---------------- problem constants (match reference) ----------------
#define NN      50000          // divisible by 16 -> GEMM has no tail
#define EE      800000
#define E2      (EE + NN)      // edges + self loops
#define NODE_IN 16
#define EDGE_IN 8
#define HID     128
#define EH      64
#define HEADS   4
#define HC      32
#define EPS     1e-5f
#define SLOPE   0.2f

typedef __attribute__((ext_vector_type(16))) __bf16 v16bf;
typedef __attribute__((ext_vector_type(8)))  float  v8f;
typedef int v4i_async __attribute__((vector_size(16)));   // matches builtin param type

// CDNA5 async global->LDS staging (ASYNCcnt path), guarded so compile never breaks
#if defined(__gfx1250__) && __has_builtin(__builtin_amdgcn_global_load_async_to_lds_b128) && \
    __has_builtin(__builtin_amdgcn_s_wait_asynccnt)
#define HAVE_ASYNC_LDS 1
#define ASYNC_CP_B128(gsrc, ldst)                                                  \
  __builtin_amdgcn_global_load_async_to_lds_b128(                                  \
      (__attribute__((address_space(1))) v4i_async*)(gsrc),                        \
      (__attribute__((address_space(3))) v4i_async*)(ldst), 0, 0)
#endif

// ---------------- helpers ----------------
__device__ __forceinline__ unsigned short f32_to_bf16(float f) {
  unsigned int u = __float_as_uint(f);
  u = (u + 0x7FFFu + ((u >> 16) & 1u)) >> 16;   // round-to-nearest-even
  return (unsigned short)u;
}

__device__ __forceinline__ unsigned int f32_to_ordkey(float f) {
  unsigned int u = __float_as_uint(f);
  return (u >> 31) ? ~u : (u | 0x80000000u);
}
__device__ __forceinline__ float ordkey_to_f32(unsigned int k) {
  unsigned int u = (k & 0x80000000u) ? (k & 0x7FFFFFFFu) : ~k;
  return __uint_as_float(u);
}

// ---------------- generic fill ----------------
__global__ void k_fill_u32(unsigned int* __restrict__ p, long long n, unsigned int v) {
  long long i = (long long)blockIdx.x * blockDim.x + threadIdx.x;
  long long stride = (long long)gridDim.x * blockDim.x;
  for (; i < n; i += stride) p[i] = v;
}

// ---------------- f32 -> bf16 convert ----------------
__global__ void k_f32_to_bf16(const float* __restrict__ in, unsigned short* __restrict__ out,
                              long long n) {
  long long i = (long long)blockIdx.x * blockDim.x + threadIdx.x;
  if (i < n) out[i] = f32_to_bf16(in[i]);
}

// ---------------- node preprocess: h = relu(LN(x@W + b)) ----------------
__global__ __launch_bounds__(HID) void k_node_pre(const float* __restrict__ x,
                                                  const float* __restrict__ w,
                                                  const float* __restrict__ b,
                                                  const float* __restrict__ g,
                                                  const float* __restrict__ be,
                                                  float* __restrict__ h) {
  __shared__ float sm[HID];
  int ch = threadIdx.x;
  long long n = blockIdx.x;
  float acc = b[ch];
  const float* xr = x + n * NODE_IN;
  #pragma unroll
  for (int k = 0; k < NODE_IN; ++k) acc += xr[k] * w[k * HID + ch];
  sm[ch] = acc; __syncthreads();
  for (int o = HID / 2; o > 0; o >>= 1) { if (ch < o) sm[ch] += sm[ch + o]; __syncthreads(); }
  float mean = sm[0] * (1.0f / HID); __syncthreads();
  float d = acc - mean;
  sm[ch] = d * d; __syncthreads();
  for (int o = HID / 2; o > 0; o >>= 1) { if (ch < o) sm[ch] += sm[ch + o]; __syncthreads(); }
  float var = sm[0] * (1.0f / HID);
  float y = d * rsqrtf(var + EPS) * g[ch] + be[ch];
  h[n * HID + ch] = fmaxf(y, 0.0f);
}

// ---------------- edge preprocess: e = relu(LN(ea@W + b)) ----------------
__global__ __launch_bounds__(EH) void k_edge_pre(const float* __restrict__ ea,
                                                 const float* __restrict__ w,
                                                 const float* __restrict__ b,
                                                 const float* __restrict__ g,
                                                 const float* __restrict__ be,
                                                 float* __restrict__ e) {
  __shared__ float sm[EH];
  int ch = threadIdx.x;
  long long j = blockIdx.x;
  float acc = b[ch];
  const float* er = ea + j * EDGE_IN;
  #pragma unroll
  for (int k = 0; k < EDGE_IN; ++k) acc += er[k] * w[k * EH + ch];
  sm[ch] = acc; __syncthreads();
  for (int o = EH / 2; o > 0; o >>= 1) { if (ch < o) sm[ch] += sm[ch + o]; __syncthreads(); }
  float mean = sm[0] * (1.0f / EH); __syncthreads();
  float d = acc - mean;
  sm[ch] = d * d; __syncthreads();
  for (int o = EH / 2; o > 0; o >>= 1) { if (ch < o) sm[ch] += sm[ch + o]; __syncthreads(); }
  float var = sm[0] * (1.0f / EH);
  float y = d * rsqrtf(var + EPS) * g[ch] + be[ch];
  e[j * EH + ch] = fmaxf(y, 0.0f);
}

// ---------------- self-loop edge attr: mean of incoming e per dst ----------------
__global__ void k_loop_accum(const float* __restrict__ e, const int* __restrict__ ei,
                             float* __restrict__ loope, float* __restrict__ deg) {
  long long t = (long long)blockIdx.x * blockDim.x + threadIdx.x;
  if (t >= (long long)EE * EH) return;
  int j = (int)(t >> 6), c = (int)(t & 63);
  int d = ei[EE + j];
  atomicAdd(&loope[(long long)d * EH + c], e[(long long)j * EH + c]);
  if (c == 0) atomicAdd(&deg[d], 1.0f);
}
__global__ void k_loop_div(float* __restrict__ loope, const float* __restrict__ deg) {
  long long t = (long long)blockIdx.x * blockDim.x + threadIdx.x;
  if (t >= (long long)NN * EH) return;
  int n = (int)(t >> 6);
  loope[t] /= fmaxf(deg[n], 1.0f);
}

// ---------------- V[layer][k][h] = sum_c We[k, h*32+c]*a_e[h,c] (64x4 per layer) ----------------
__global__ __launch_bounds__(256) void k_compute_V(const float* __restrict__ ew,
                                                   const float* __restrict__ ae,
                                                   float* __restrict__ V) {
  int layer = blockIdx.x;
  int hh = threadIdx.x & 3, k = threadIdx.x >> 2;   // 64*4 = 256 threads
  float a = 0.0f;
  const float* ewp = ew + (long long)layer * EH * HID + (long long)k * HID + hh * HC;
  const float* aep = ae + (long long)layer * HID + hh * HC;
  #pragma unroll
  for (int c = 0; c < HC; ++c) a += ewp[c] * aep[c];
  V[layer * (EH * HEADS) + k * HEADS + hh] = a;
}

// ---------------- per-edge attention edge-term: eA[L][j][h] = feat(j) . V[L][:,h] ----------------
__global__ void k_edge_alpha_e(const float* __restrict__ e, const float* __restrict__ loope,
                               const float* __restrict__ V, float* __restrict__ eA, int layer) {
  long long t = (long long)blockIdx.x * blockDim.x + threadIdx.x;
  if (t >= (long long)E2 * HEADS) return;
  int j = (int)(t >> 2), hh = (int)(t & 3);
  const float* feat = (j < EE) ? (e + (long long)j * EH) : (loope + (long long)(j - EE) * EH);
  const float* vp = V + layer * (EH * HEADS);
  float a = 0.0f;
  #pragma unroll
  for (int k = 0; k < EH; ++k) a += feat[k] * vp[k * HEADS + hh];
  eA[(long long)layer * E2 * HEADS + t] = a;
}

// ---------------- pre-swizzle B (128x128 f32 -> bf16, fragment-contiguous) ----------------
// dest index i -> (t = ntile, kc = K-chunk, l = lane, e = element)
//   i = ((t*4 + kc)*32 + l)*16 + e
//   kb = kc*32 + (l>>4)*8 + (e&7) + (e>>3)*16  (WMMA B 32x16 layout), col = t*16 + (l&15)
__global__ void k_swizzleB(const float* __restrict__ W, unsigned short* __restrict__ Bsw) {
  int i = blockIdx.x * blockDim.x + threadIdx.x;
  if (i >= 8 * 4 * 32 * 16) return;
  int e = i & 15, l = (i >> 4) & 31, kc = (i >> 9) & 3, t = i >> 11;
  int kb = kc * 32 + (l >> 4) * 8 + (e & 7) + (e >> 3) * 16;
  int col = t * 16 + (l & 15);
  Bsw[i] = f32_to_bf16(W[kb * HID + col]);
}

// ---------------- WMMA GEMM: C[M x 128] = A[M x 128] @ B[128 x 128] ----------------
// A row-major bf16; B pre-swizzled by k_swizzleB. M must be a multiple of 16.
__global__ __launch_bounds__(256) void k_wmma_gemm(const unsigned short* __restrict__ A,
                                                   const unsigned short* __restrict__ Bsw,
                                                   float* __restrict__ C, int M) {
  __shared__ __align__(16) unsigned short sA[16 * HID];        // 4 KB, contiguous tile
  __shared__ __align__(16) unsigned short sB[8 * 4 * 32 * 16]; // 32 KB, fragment-major
  int tid = threadIdx.x;
  long long row0 = (long long)blockIdx.x * 16;
  const char* gA = (const char*)(A + row0 * HID);   // 4096 B contiguous
  const char* gB = (const char*)Bsw;                // 32768 B contiguous

#ifdef HAVE_ASYNC_LDS
  // bulk tile staging through the async global->LDS path (no VGPR round-trip)
  {
    char* lA = (char*)sA;
    char* lB = (char*)sB;
    // A: 256 x b128
    ASYNC_CP_B128(gA + tid * 16, lA + tid * 16);
    // B: 2048 x b128 -> 8 per thread
    #pragma unroll
    for (int i = 0; i < 8; ++i)
      ASYNC_CP_B128(gB + (tid + i * 256) * 16, lB + (tid + i * 256) * 16);
    __builtin_amdgcn_s_wait_asynccnt(0);
  }
  __syncthreads();
#else
  {
    const uint4* gA4 = (const uint4*)gA;
    const uint4* gB4 = (const uint4*)gB;
    uint4* lA4 = (uint4*)sA;
    uint4* lB4 = (uint4*)sB;
    lA4[tid] = gA4[tid];
    #pragma unroll
    for (int i = 0; i < 8; ++i) lB4[tid + i * 256] = gB4[tid + i * 256];
  }
  __syncthreads();
#endif

  int lane = tid & 31, wv = tid >> 5;               // 8 waves -> 8 N-tiles of 16 cols
  int half = lane >> 4;
  int m = lane & 15, ncol = lane & 15;

  v8f acc = {};
  #pragma unroll
  for (int kc = 0; kc < 4; ++kc) {
    int k0 = kc * 32;
    union { v16bf v; uint4 q[2]; } fa, fb;
    // A 16x32 bf16 fragment: two aligned 16B runs
    const uint4* pA = (const uint4*)(sA + m * HID + k0 + half * 8);
    fa.q[0] = pA[0];
    fa.q[1] = pA[2];   // +16 ushorts = +32 B
    // B fragment: one aligned 32B run (pre-swizzled)
    const uint4* pB = (const uint4*)(sB + (((wv * 4 + kc) * 32 + lane) << 4));
    fb.q[0] = pB[0];
    fb.q[1] = pB[1];
    acc = __builtin_amdgcn_wmma_f32_16x16x32_bf16(false, fa.v, false, fb.v,
                                                  (short)0, acc, false, false);
  }
  // D layout: n = lane&15, m = r + 8*(lane>=16)
  #pragma unroll
  for (int r = 0; r < 8; ++r) {
    long long row = row0 + half * 8 + r;
    C[row * HID + wv * 16 + ncol] = acc[r];
  }
}

// ---------------- per-node attention coefs: al_s, al_d ----------------
__global__ __launch_bounds__(HID) void k_attn_coef(const float* __restrict__ xs,
                                                   const float* __restrict__ as_,
                                                   const float* __restrict__ ad_,
                                                   float* __restrict__ als,
                                                   float* __restrict__ ald) {
  long long n = blockIdx.x;
  int tid = threadIdx.x, lane = tid & 31, wv = tid >> 5;   // head = wave
  float xv = xs[n * HID + tid];
  float vs = xv * as_[tid];
  float vd = xv * ad_[tid];
  #pragma unroll
  for (int o = 16; o > 0; o >>= 1) {
    vs += __shfl_down(vs, o, 32);
    vd += __shfl_down(vd, o, 32);
  }
  if (lane == 0) { als[n * HEADS + wv] = vs; ald[n * HEADS + wv] = vd; }
}

// ---------------- edge pass 1: alpha + leaky relu + segment max ----------------
__global__ void k_alpha_max(const float* __restrict__ als, const float* __restrict__ ald,
                            const float* __restrict__ eA_l, const int* __restrict__ ei,
                            float* __restrict__ alphabuf, unsigned int* __restrict__ amax) {
  long long t = (long long)blockIdx.x * blockDim.x + threadIdx.x;
  if (t >= (long long)E2 * HEADS) return;
  int j = (int)(t >> 2), hh = (int)(t & 3);
  int s = (j < EE) ? ei[j] : (j - EE);
  int d = (j < EE) ? ei[EE + j] : (j - EE);
  float a = als[s * HEADS + hh] + ald[d * HEADS + hh] + eA_l[t];
  a = (a >= 0.0f) ? a : SLOPE * a;
  alphabuf[t] = a;
  atomicMax(&amax[d * HEADS + hh], f32_to_ordkey(a));
}

// ---------------- edge pass 2: exp + segment sum ----------------
__global__ void k_alpha_exp(const int* __restrict__ ei, const unsigned int* __restrict__ amax,
                            float* __restrict__ alphabuf, float* __restrict__ asum) {
  long long t = (long long)blockIdx.x * blockDim.x + threadIdx.x;
  if (t >= (long long)E2 * HEADS) return;
  int j = (int)(t >> 2), hh = (int)(t & 3);
  int d = (j < EE) ? ei[EE + j] : (j - EE);
  float m = ordkey_to_f32(amax[d * HEADS + hh]);
  float ex = expf(alphabuf[t] - m);
  alphabuf[t] = ex;
  atomicAdd(&asum[d * HEADS + hh], ex);
}

// ---------------- edge pass 3: weighted message scatter ----------------
__global__ void k_scatter(const int* __restrict__ ei, const float* __restrict__ alphabuf,
                          const float* __restrict__ asum, const float* __restrict__ xs,
                          float* __restrict__ agg) {
  long long t = (long long)blockIdx.x * blockDim.x + threadIdx.x;
  if (t >= (long long)E2 * HID) return;
  int j = (int)(t >> 7), ch = (int)(t & 127), hh = ch >> 5;
  int s = (j < EE) ? ei[j] : (j - EE);
  int d = (j < EE) ? ei[EE + j] : (j - EE);
  float w = alphabuf[(long long)j * HEADS + hh] / asum[d * HEADS + hh];
  atomicAdd(&agg[(long long)d * HID + ch], xs[(long long)s * HID + ch] * w);
}

// ---------------- bias add + BN column stats ----------------
__global__ __launch_bounds__(HID) void k_bias_stats(float* __restrict__ agg,
                                                    const float* __restrict__ gb,
                                                    float* __restrict__ colsum,
                                                    float* __restrict__ colsq) {
  int ch = threadIdx.x;
  float bb = gb[ch];
  float s = 0.0f, q = 0.0f;
  for (long long n = blockIdx.x; n < NN; n += gridDim.x) {
    float v = agg[n * HID + ch] + bb;
    agg[n * HID + ch] = v;
    s += v; q += v * v;
  }
  atomicAdd(&colsum[ch], s);
  atomicAdd(&colsq[ch], q);
}

// ---------------- BN apply + residual + relu -> h ----------------
__global__ void k_bn_apply(float* __restrict__ h, const float* __restrict__ agg,
                           const float* __restrict__ colsum, const float* __restrict__ colsq,
                           const float* __restrict__ bng, const float* __restrict__ bnb) {
  long long i = (long long)blockIdx.x * blockDim.x + threadIdx.x;
  if (i >= (long long)NN * HID) return;
  int ch = (int)(i & 127);
  float m = colsum[ch] * (1.0f / NN);
  float var = colsq[ch] * (1.0f / NN) - m * m;
  float y = (agg[i] - m) * rsqrtf(var + EPS) * bng[ch] + bnb[ch] + h[i];
  h[i] = fmaxf(y, 0.0f);
}

// ---------------- final LN: out = LN(xs + fp_b) ----------------
__global__ __launch_bounds__(HID) void k_final_ln(const float* __restrict__ xs,
                                                  const float* __restrict__ fb,
                                                  const float* __restrict__ fg,
                                                  const float* __restrict__ fbe,
                                                  float* __restrict__ out) {
  __shared__ float sm[HID];
  int ch = threadIdx.x;
  long long n = blockIdx.x;
  float acc = xs[n * HID + ch] + fb[ch];
  sm[ch] = acc; __syncthreads();
  for (int o = HID / 2; o > 0; o >>= 1) { if (ch < o) sm[ch] += sm[ch + o]; __syncthreads(); }
  float mean = sm[0] * (1.0f / HID); __syncthreads();
  float d = acc - mean;
  sm[ch] = d * d; __syncthreads();
  for (int o = HID / 2; o > 0; o >>= 1) { if (ch < o) sm[ch] += sm[ch + o]; __syncthreads(); }
  float var = sm[0] * (1.0f / HID);
  out[n * HID + ch] = d * rsqrtf(var + EPS) * fg[ch] + fbe[ch];
}

// ============================ host launcher ============================
static inline unsigned blocks_for(long long n, int b) { return (unsigned)((n + b - 1) / b); }

extern "C" void kernel_launch(void* const* d_in, const int* in_sizes, int n_in,
                              void* d_out, int out_size, void* d_ws, size_t ws_size,
                              hipStream_t stream) {
  const float* x      = (const float*)d_in[0];
  const int*   ei     = (const int*)  d_in[1];
  const float* eattr  = (const float*)d_in[2];
  const float* np_w   = (const float*)d_in[3];
  const float* np_b   = (const float*)d_in[4];
  const float* np_g   = (const float*)d_in[5];
  const float* np_be  = (const float*)d_in[6];
  const float* ep_w   = (const float*)d_in[7];
  const float* ep_b   = (const float*)d_in[8];
  const float* ep_g   = (const float*)d_in[9];
  const float* ep_be  = (const float*)d_in[10];
  const float* gat_w  = (const float*)d_in[11];
  const float* gat_as = (const float*)d_in[12];
  const float* gat_ad = (const float*)d_in[13];
  const float* gat_ew = (const float*)d_in[14];
  const float* gat_ae = (const float*)d_in[15];
  const float* gat_b  = (const float*)d_in[16];
  const float* bn_g   = (const float*)d_in[17];
  const float* bn_b   = (const float*)d_in[18];
  const float* fp_w   = (const float*)d_in[19];
  const float* fp_b   = (const float*)d_in[20];
  const float* fp_g   = (const float*)d_in[21];
  const float* fp_be  = (const float*)d_in[22];
  float* out = (float*)d_out;

  // ---- workspace carve-out ----
  char* ws = (char*)d_ws;
  size_t off = 0;
  auto carve = [&](size_t bytes) -> char* {
    char* p = ws + off;
    off += (bytes + 255) & ~(size_t)255;
    return p;
  };
  float*          h      = (float*)         carve((size_t)NN * HID * 4);
  unsigned short* hb     = (unsigned short*)carve((size_t)NN * HID * 2);
  unsigned short* Bsw    = (unsigned short*)carve((size_t)HID * HID * 2);
  float*          loope  = (float*)         carve((size_t)NN * EH * 4);
  float*          deg    = (float*)         carve((size_t)NN * 4);
  float*          als    = (float*)         carve((size_t)NN * HEADS * 4);
  float*          ald    = (float*)         carve((size_t)NN * HEADS * 4);
  unsigned int*   amax   = (unsigned int*)  carve((size_t)NN * HEADS * 4);
  float*          asum   = (float*)         carve((size_t)NN * HEADS * 4);
  float*          eA     = (float*)         carve((size_t)3 * E2 * HEADS * 4);
  float*          Vmat   = (float*)         carve((size_t)3 * EH * HEADS * 4);
  float*          colsum = (float*)         carve((size_t)HID * 4);
  float*          colsq  = (float*)         carve((size_t)HID * 4);
  // big region: e (E x 64) alive only in prologue; reuse for xs / agg / alpha in layers
  char*  bigreg = carve((size_t)EE * EH * 4);
  float* efeat  = (float*)bigreg;
  float* xs     = (float*)bigreg;                                  // N*128 f32
  float* agg    = (float*)(bigreg + (size_t)NN * HID * 4);         // N*128 f32
  float* alphab = (float*)(bigreg + (size_t)2 * NN * HID * 4);     // E2*4  f32

  const int B = 256;

  // ---- prologue ----
  k_node_pre<<<NN, HID, 0, stream>>>(x, np_w, np_b, np_g, np_be, h);
  k_edge_pre<<<EE, EH, 0, stream>>>(eattr, ep_w, ep_b, ep_g, ep_be, efeat);

  k_fill_u32<<<1024, B, 0, stream>>>((unsigned int*)loope, (long long)NN * EH, 0u);
  k_fill_u32<<<256, B, 0, stream>>>((unsigned int*)deg, NN, 0u);
  k_loop_accum<<<blocks_for((long long)EE * EH, B), B, 0, stream>>>(efeat, ei, loope, deg);
  k_loop_div<<<blocks_for((long long)NN * EH, B), B, 0, stream>>>(loope, deg);

  k_compute_V<<<3, 256, 0, stream>>>(gat_ew, gat_ae, Vmat);
  for (int L = 0; L < 3; ++L)
    k_edge_alpha_e<<<blocks_for((long long)E2 * HEADS, B), B, 0, stream>>>(
        efeat, loope, Vmat, eA, L);

  // ---- 3 GAT layers ----
  for (int L = 0; L < 3; ++L) {
    k_f32_to_bf16<<<blocks_for((long long)NN * HID, B), B, 0, stream>>>(
        h, hb, (long long)NN * HID);
    k_swizzleB<<<64, B, 0, stream>>>(gat_w + (long long)L * HID * HID, Bsw);
    k_wmma_gemm<<<NN / 16, 256, 0, stream>>>(hb, Bsw, xs, NN);

    k_attn_coef<<<NN, HID, 0, stream>>>(xs, gat_as + (long long)L * HID,
                                        gat_ad + (long long)L * HID, als, ald);

    k_fill_u32<<<64, B, 0, stream>>>(amax, (long long)NN * HEADS, 0u);      // 0 sorts below ordkey(-inf)
    k_fill_u32<<<64, B, 0, stream>>>((unsigned int*)asum, (long long)NN * HEADS, 0u);
    k_fill_u32<<<4096, B, 0, stream>>>((unsigned int*)agg, (long long)NN * HID, 0u);
    k_fill_u32<<<1, B, 0, stream>>>((unsigned int*)colsum, HID, 0u);
    k_fill_u32<<<1, B, 0, stream>>>((unsigned int*)colsq, HID, 0u);

    const float* eA_l = eA + (long long)L * E2 * HEADS;
    k_alpha_max<<<blocks_for((long long)E2 * HEADS, B), B, 0, stream>>>(
        als, ald, eA_l, ei, alphab, amax);
    k_alpha_exp<<<blocks_for((long long)E2 * HEADS, B), B, 0, stream>>>(
        ei, amax, alphab, asum);
    k_scatter<<<blocks_for((long long)E2 * HID, B), B, 0, stream>>>(
        ei, alphab, asum, xs, agg);

    k_bias_stats<<<512, HID, 0, stream>>>(agg, gat_b + (long long)L * HID, colsum, colsq);
    k_bn_apply<<<blocks_for((long long)NN * HID, B), B, 0, stream>>>(
        h, agg, colsum, colsq, bn_g + (long long)L * HID, bn_b + (long long)L * HID);
  }

  // ---- final projection + LN ----
  k_f32_to_bf16<<<blocks_for((long long)NN * HID, B), B, 0, stream>>>(
      h, hb, (long long)NN * HID);
  k_swizzleB<<<64, B, 0, stream>>>(fp_w, Bsw);
  k_wmma_gemm<<<NN / 16, 256, 0, stream>>>(hb, Bsw, xs, NN);
  k_final_ln<<<NN, HID, 0, stream>>>(xs, fp_b, fp_g, fp_be, out);

  (void)in_sizes; (void)n_in; (void)out_size; (void)ws_size;
}
